// DualLSTM_25692494364907
// MI455X (gfx1250) — matrix operand order repaired and checked by reading, not compiled, not run on hardware
//
#include <hip/hip_runtime.h>
#include <hip/hip_bf16.h>
#include <cstdint>
#include <cstddef>

// ---------------- problem constants ----------------
#define T1     4096      // T-1 timesteps
#define EDIM   1024
#define HDIM   1024
#define D2     2048      // 2*H
#define KDIM   2048
#define NVOCAB 50257

// ---------------- CDNA5 types ----------------
typedef __attribute__((ext_vector_type(16))) __bf16        v16bf;
typedef __attribute__((ext_vector_type(8)))  float         v8f;
typedef __attribute__((ext_vector_type(4)))  unsigned int  u32x4;
typedef __attribute__((ext_vector_type(8)))  unsigned int  u32x8;
typedef __attribute__((ext_vector_type(2)))  unsigned int  u32x2;
typedef __attribute__((ext_vector_type(4)))  int           i32x4;

struct Frag32 { u32x4 lo; u32x4 hi; };   // 32B, trivially copyable

static __device__ __forceinline__ v16bf frag_cast(const Frag32& f) {
    return __builtin_bit_cast(v16bf, f);
}

static __device__ __forceinline__ unsigned short f2bf(float f) {
    unsigned u = __float_as_uint(f);
    unsigned r = u + 0x7fffu + ((u >> 16) & 1u);   // RNE
    return (unsigned short)(r >> 16);
}

static __device__ __forceinline__ float sigf(float x) {
    return 1.f / (1.f + __expf(-x));
}
static __device__ __forceinline__ float tanhfast(float x) {
    return 1.f - 2.f / (__expf(2.f * x) + 1.f);
}

// low 32 bits of a generic pointer to LDS == LDS byte address (ISA §10.2 aperture)
static __device__ __forceinline__ unsigned lds_lo32(const void* p) {
    return (unsigned)(unsigned long long)(size_t)p;
}

// ---------------- Tensor Data Mover: 2D tile load -> LDS ----------------
// D# per CDNA5 ISA.  bf16 elements (data_size=1),
// tile = tile0 x tile1 rows, row stride = stride_elems, LDS padding:
// pad_interval=3 (every 16 DWORDs = 64B row) , pad_amount=3 (4 DWORDs = 16B)
// -> LDS row pitch 80B, matching LDSS=40 ushorts below.
static __device__ __forceinline__ void tdm_load_2d_bf16(
    const void* gptr, unsigned lds_byte,
    unsigned tdim0, unsigned tdim1,      // remaining tensor extent from tile origin (OOB -> 0)
    unsigned tile0, unsigned tile1,      // tile dims (elems, rows)
    unsigned stride_elems)
{
    unsigned long long ga = (unsigned long long)(size_t)gptr;
    u32x4 g0;
    g0[0] = 1u;                                              // count=1, user mode
    g0[1] = lds_byte;                                        // lds_addr
    g0[2] = (unsigned)ga;                                    // global_addr[31:0]
    g0[3] = (unsigned)((ga >> 32) & 0x01FFFFFFull)           // global_addr[56:32]
          | (2u << 30);                                      // type=2 ("image")
    u32x8 g1;
    g1[0] = (1u << 16)        // data_size = 1 (2 bytes)
          | (1u << 20)        // pad_enable
          | (3u << 22)        // pad_interval: every 16 DWORDs
          | (3u << 25);       // pad_amount: 4 DWORDs
    g1[1] = (tdim0 & 0xFFFFu) << 16;                         // tensor_dim0[15:0]
    g1[2] = (tdim0 >> 16) | ((tdim1 & 0xFFFFu) << 16);       // dim0 hi | dim1 lo
    g1[3] = (tdim1 >> 16) | (tile0 << 16);                   // dim1 hi | tile_dim0
    g1[4] = tile1;                                           // tile_dim1 (tile_dim2=0)
    g1[5] = stride_elems;                                    // tensor_dim0_stride lo32
    g1[6] = 0u;                                              // stride hi | dim1_stride lo
    g1[7] = 0u;
    asm volatile("tensor_load_to_lds %0, %1" :: "s"(g0), "s"(g1) : "memory");
}

// ---------------- async global->LDS helpers (guarded) ----------------
#if __has_builtin(__builtin_amdgcn_global_load_async_to_lds_b128)
#define HAVE_ASYNC_LDS 1
#endif

static __device__ __forceinline__ void wait_asynccnt0() {
#if __has_builtin(__builtin_amdgcn_s_wait_asynccnt)
    __builtin_amdgcn_s_wait_asynccnt(0);
#else
    asm volatile("s_wait_asynccnt 0" ::: "memory");
#endif
}

#ifdef HAVE_ASYNC_LDS
static __device__ __forceinline__ void async_copy16(const void* g, void* lds) {
    __builtin_amdgcn_global_load_async_to_lds_b128(
        (__attribute__((address_space(1))) i32x4*)(size_t)g,
        (__attribute__((address_space(3))) i32x4*)lds, 0, 0);
}
#endif

// ---------------- workspace layout (bytes) ----------------
static constexpr size_t OFF_XSEQ  = 0;                                   // 4096*1024 f32
static constexpr size_t OFF_LSTMF = OFF_XSEQ  + (size_t)T1*EDIM*4;       // 4096*2048 f32
static constexpr size_t OFF_LSTMB = OFF_LSTMF + (size_t)T1*D2*4;         // 4096*2048 bf16
static constexpr size_t OFF_H1B   = OFF_LSTMB + (size_t)T1*D2*2;         // 4096*2048 bf16
static constexpr size_t OFF_W1B   = OFF_H1B   + (size_t)T1*D2*2;         // 2048*2048 bf16
static constexpr size_t OFF_W2B   = OFF_W1B   + (size_t)D2*KDIM*2;       // 50257*2048 bf16
static constexpr size_t OFF_HEN   = OFF_W2B   + (size_t)NVOCAB*KDIM*2;   // 1024 f32
static constexpr size_t OFF_HCN   = OFF_HEN   + 4096;
static constexpr size_t OFF_BAR   = OFF_HCN   + 4096;

// ---------------- barrier / init ----------------
__global__ void init_bar_kernel(unsigned* bar) {
    if (threadIdx.x == 0 && blockIdx.x == 0) *bar = 0u;
}

#define SCAN_NBLK 128
#define SCAN_TPB  256   // 8 wave32 per block -> 1024 waves == HDIM cells

static __device__ __forceinline__ void grid_sync(unsigned* bar, unsigned epoch) {
    __threadfence();
    __syncthreads();
    if (threadIdx.x == 0) {
        __hip_atomic_fetch_add(bar, 1u, __ATOMIC_ACQ_REL, __HIP_MEMORY_SCOPE_AGENT);
        const unsigned target = epoch * (unsigned)SCAN_NBLK;
        while (__hip_atomic_load(bar, __ATOMIC_ACQUIRE, __HIP_MEMORY_SCOPE_AGENT) < target)
            __builtin_amdgcn_s_sleep(2);
    }
    __syncthreads();
    __threadfence();
}

// ---------------- embedding gather ----------------
__global__ void embed_kernel(const int* __restrict__ sent,
                             const float* __restrict__ emb,
                             float* __restrict__ xs) {
    const int t   = blockIdx.x;
    const int row = sent[t];
    const float4* src = (const float4*)(emb + (size_t)row * EDIM);
    float4*       dst = (float4*)(xs + (size_t)t * EDIM);
    for (int i = threadIdx.x; i < EDIM / 4; i += blockDim.x) dst[i] = src[i];
}

// ---------------- f32 -> bf16 bulk convert (n divisible by 4) ----------------
__global__ void cvt_bf16_kernel(const float* __restrict__ in,
                                unsigned short* __restrict__ out,
                                size_t n4) {
    size_t i      = (size_t)blockIdx.x * blockDim.x + threadIdx.x;
    size_t stride = (size_t)gridDim.x * blockDim.x;
    for (; i < n4; i += stride) {
        __builtin_prefetch(in + (i + stride) * 4, 0, 1);   // global_prefetch_b8
        float4 v = ((const float4*)in)[i];
        unsigned lo = (unsigned)f2bf(v.x) | ((unsigned)f2bf(v.y) << 16);
        unsigned hi = (unsigned)f2bf(v.z) | ((unsigned)f2bf(v.w) << 16);
        u32x2 p; p.x = lo; p.y = hi;
        ((u32x2*)out)[i] = p;
    }
}

// ---------------- persistent sequential LSTM scan ----------------
// 1024 waves total; wave w owns cell unit j = w. Cell state c lives in
// registers for the whole 4096-step scan. Two device-scope barriers/step.
__global__ void __launch_bounds__(SCAN_TPB)
lstm_scan_kernel(const float* __restrict__ xseq,
                 const int*   __restrict__ lang_ids,
                 const float* __restrict__ Wih_en, const float* __restrict__ Whh_en,
                 const float* __restrict__ Wih_cn, const float* __restrict__ Whh_cn,
                 float* __restrict__ h_en, float* __restrict__ h_cn,
                 float* __restrict__ lstm_f32, unsigned short* __restrict__ lstm_bf,
                 unsigned* __restrict__ bar) {
    __shared__ float sh_x[EDIM];
    __shared__ float sh_h[HDIM];
    const int tid  = threadIdx.x;
    const int lane = tid & 31;
    const int wave = tid >> 5;
    const int j    = blockIdx.x * (SCAN_TPB / 32) + wave;  // 0..1023

    unsigned epoch = 0;
    float c = 0.f;
    if (lane == 0) { h_en[j] = 0.f; h_cn[j] = 0.f; }
    epoch++; grid_sync(bar, epoch);

    for (int t = 0; t < T1; ++t) {
        const bool m = lang_ids[t] > 0;
        const float* Wih  = m ? Wih_en : Wih_cn;
        const float* WhhA = m ? Whh_en : Whh_cn;
        const float* WhhO = m ? Whh_cn : Whh_en;
        float* hA = m ? h_en : h_cn;
        float* hO = m ? h_cn : h_en;
        const int offA = m ? 0 : HDIM;
        const int offO = m ? HDIM : 0;

        // ---- phase A: active cell gets x_t; h = hA (previous) ----
#ifdef HAVE_ASYNC_LDS
        {
            const int i4 = tid * 4;          // 256 lanes * 4 f32 = 1024
            async_copy16(xseq + (size_t)t * EDIM + i4, &sh_x[i4]);
            async_copy16(hA + i4, &sh_h[i4]);
            wait_asynccnt0();
        }
#else
        for (int i = tid; i < EDIM; i += SCAN_TPB) sh_x[i] = xseq[(size_t)t * EDIM + i];
        for (int i = tid; i < HDIM; i += SCAN_TPB) sh_h[i] = hA[i];
#endif
        __syncthreads();

        float acc[4];
        #pragma unroll
        for (int g = 0; g < 4; ++g) {
            const float* w1 = Wih  + (size_t)(g * HDIM + j) * EDIM;
            const float* w2 = WhhA + (size_t)(g * HDIM + j) * HDIM;
            float s = 0.f;
            #pragma unroll 8
            for (int k = lane; k < EDIM; k += 32)
                s = __fmaf_rn(w1[k], sh_x[k], __fmaf_rn(w2[k], sh_h[k], s));
            acc[g] = s;
        }
        #pragma unroll
        for (int g = 0; g < 4; ++g)
            #pragma unroll
            for (int d = 16; d > 0; d >>= 1) acc[g] += __shfl_xor(acc[g], d, 32);

        c = sigf(acc[1]) * c + sigf(acc[0]) * tanhfast(acc[2]);
        float hAn = sigf(acc[3]) * tanhfast(c);
        if (lane == 0) {
            hA[j] = hAn;
            lstm_f32[(size_t)t * D2 + offA + j] = hAn;
            lstm_bf [(size_t)t * D2 + offA + j] = f2bf(hAn);
        }
        epoch++; grid_sync(bar, epoch);

        // ---- phase B: other cell, dummy x (zero), h = fresh hA ----
#ifdef HAVE_ASYNC_LDS
        {
            const int i4 = tid * 4;
            async_copy16(hA + i4, &sh_h[i4]);
            wait_asynccnt0();
        }
#else
        for (int i = tid; i < HDIM; i += SCAN_TPB) sh_h[i] = hA[i];
#endif
        __syncthreads();

        #pragma unroll
        for (int g = 0; g < 4; ++g) {
            const float* w2 = WhhO + (size_t)(g * HDIM + j) * HDIM;
            float s = 0.f;
            #pragma unroll 8
            for (int k = lane; k < HDIM; k += 32)
                s = __fmaf_rn(w2[k], sh_h[k], s);
            acc[g] = s;
        }
        #pragma unroll
        for (int g = 0; g < 4; ++g)
            #pragma unroll
            for (int d = 16; d > 0; d >>= 1) acc[g] += __shfl_xor(acc[g], d, 32);

        c = sigf(acc[1]) * c + sigf(acc[0]) * tanhfast(acc[2]);
        float hOn = sigf(acc[3]) * tanhfast(c);
        if (lane == 0) {
            hO[j] = hOn;
            lstm_f32[(size_t)t * D2 + offO + j] = hOn;
            lstm_bf [(size_t)t * D2 + offO + j] = f2bf(hOn);
        }
        epoch++; grid_sync(bar, epoch);
    }
}

// ---------------- WMMA bf16 GEMM:  C[M,N] = A[M,K] * W[N,K]^T + bias ----------------
// Tiles staged into LDS by the Tensor Data Mover (one D# per tile, wave 0).
#define BM 128
#define BN 128
#define BK 32
#define LDSS 40   // ushort row pitch (80B = 64B data + 16B TDM pad)

template <bool RELU, bool OUTBF>
__global__ void __launch_bounds__(256)
gemm_bf16_kernel(const unsigned short* __restrict__ A,
                 const unsigned short* __restrict__ W,
                 const float* __restrict__ bias,
                 float* __restrict__ outF,
                 unsigned short* __restrict__ outBF,
                 int M, int N, int K) {
    __shared__ unsigned short As[BM * LDSS];
    __shared__ unsigned short Bs[BN * LDSS];

    const int tid  = threadIdx.x;
    const int lane = tid & 31;
    const int wave = tid >> 5;     // 0..7
    const int wm   = wave >> 2;    // 0..1 : 64-row band
    const int wn   = wave & 3;     // 0..3 : 32-col band
    const int m0   = blockIdx.y * BM;
    const int n0   = blockIdx.x * BN;
    const int half = lane >> 4;    // K-half selector (this code's contiguous k = 16*half + s)
    const int l16  = lane & 15;

    v8f acc[4][2] = {};

    const unsigned ldsA = lds_lo32(As);
    const unsigned ldsB = lds_lo32(Bs);

    for (int kb = 0; kb < K; kb += BK) {
        __syncthreads();                       // previous tile fully consumed
        if (wave == 0) {
            // TDM: 128x32 bf16 tiles; hardware zero-fills OOB rows (fc2 N-tail)
            tdm_load_2d_bf16(A + (size_t)m0 * K + kb, ldsA,
                             (unsigned)(K - kb), (unsigned)(M - m0),
                             BK, BM, (unsigned)K);
            tdm_load_2d_bf16(W + (size_t)n0 * K + kb, ldsB,
                             (unsigned)(K - kb), (unsigned)(N - n0),
                             BK, BN, (unsigned)K);
            __builtin_amdgcn_s_wait_tensorcnt(0);
        }
        __syncthreads();                       // LDS tiles visible to all waves

        // B fragments: lane = (N=l16, K half) ; 16 contiguous K at 16*half
        Frag32 bfr[2];
        #pragma unroll
        for (int ni = 0; ni < 2; ++ni) {
            const int rb = wn * 32 + ni * 16 + l16;
            bfr[ni].lo = *((const u32x4*)&Bs[rb * LDSS + 16 * half]);
            bfr[ni].hi = *((const u32x4*)&Bs[rb * LDSS + 16 * half + 8]);
        }
        #pragma unroll
        for (int mi = 0; mi < 4; ++mi) {
            // A fragment: lane = (M=l16), 16 contiguous K at 16*half
            const int ra = wm * 64 + mi * 16 + l16;
            Frag32 af;
            af.lo = *((const u32x4*)&As[ra * LDSS + 16 * half]);
            af.hi = *((const u32x4*)&As[ra * LDSS + 16 * half + 8]);
            v16bf av = frag_cast(af);
            #pragma unroll
            for (int ni = 0; ni < 2; ++ni) {
                acc[mi][ni] = __builtin_amdgcn_wmma_f32_16x16x32_bf16(
                    false, av, false, frag_cast(bfr[ni]),
                    (short)0, acc[mi][ni], false, false);
            }
        }
    }

    // epilogue: D row = tile_m + 8*half + e, col = tile_n + l16
    #pragma unroll
    for (int mi = 0; mi < 4; ++mi) {
        #pragma unroll
        for (int ni = 0; ni < 2; ++ni) {
            const int col = n0 + wn * 32 + ni * 16 + l16;
            if (col < N) {
                const float bv = bias[col];
                #pragma unroll
                for (int e = 0; e < 8; ++e) {
                    const int row = m0 + wm * 64 + mi * 16 + half * 8 + e;
                    float v = acc[mi][ni][e] + bv;
                    if (RELU) v = v > 0.f ? v : 0.f;
                    if (OUTBF) outBF[(size_t)row * N + col] = f2bf(v);
                    else       outF [(size_t)row * N + col] = v;
                }
            }
        }
    }
}

// ---------------- tiny language head: (4096 x 2048) @ (2 x 2048)^T ----------------
__global__ void lang_head_kernel(const float* __restrict__ lstm,
                                 const float* __restrict__ lw,
                                 const float* __restrict__ lb,
                                 float* __restrict__ out) {
    const int wave = (int)((blockIdx.x * blockDim.x + threadIdx.x) >> 5);
    const int lane = threadIdx.x & 31;
    if (wave >= T1) return;
    const float* x = lstm + (size_t)wave * D2;
    float s0 = 0.f, s1 = 0.f;
    for (int k = lane; k < D2; k += 32) {
        const float xv = x[k];
        s0 = __fmaf_rn(xv, lw[k], s0);
        s1 = __fmaf_rn(xv, lw[D2 + k], s1);
    }
    #pragma unroll
    for (int d = 16; d > 0; d >>= 1) {
        s0 += __shfl_xor(s0, d, 32);
        s1 += __shfl_xor(s1, d, 32);
    }
    if (lane == 0) {
        out[(size_t)wave * 2 + 0] = s0 + lb[0];
        out[(size_t)wave * 2 + 1] = s1 + lb[1];
    }
}

// ---------------- launcher ----------------
extern "C" void kernel_launch(void* const* d_in, const int* in_sizes, int n_in,
                              void* d_out, int out_size, void* d_ws, size_t ws_size,
                              hipStream_t stream) {
    (void)in_sizes; (void)n_in; (void)out_size; (void)ws_size;

    const int*   sentence  = (const int*)  d_in[0];
    const int*   lang_ids  = (const int*)  d_in[1];
    const float* embedding = (const float*)d_in[2];
    const float* Wih_en    = (const float*)d_in[3];
    const float* Whh_en    = (const float*)d_in[4];
    const float* Wih_cn    = (const float*)d_in[5];
    const float* Whh_cn    = (const float*)d_in[6];
    const float* fc1_w     = (const float*)d_in[7];
    const float* fc1_b     = (const float*)d_in[8];
    const float* fc2_w     = (const float*)d_in[9];
    const float* fc2_b     = (const float*)d_in[10];
    const float* lang_w    = (const float*)d_in[11];
    const float* lang_b    = (const float*)d_in[12];

    char* ws = (char*)d_ws;
    float*          xseq    = (float*)         (ws + OFF_XSEQ);
    float*          lstmF   = (float*)         (ws + OFF_LSTMF);
    unsigned short* lstmB   = (unsigned short*)(ws + OFF_LSTMB);
    unsigned short* h1B     = (unsigned short*)(ws + OFF_H1B);
    unsigned short* w1B     = (unsigned short*)(ws + OFF_W1B);
    unsigned short* w2B     = (unsigned short*)(ws + OFF_W2B);
    float*          h_en    = (float*)         (ws + OFF_HEN);
    float*          h_cn    = (float*)         (ws + OFF_HCN);
    unsigned*       bar     = (unsigned*)      (ws + OFF_BAR);

    float* pred = (float*)d_out;                       // 4096 x 50257
    float* lang = pred + (size_t)T1 * NVOCAB;          // 4096 x 2

    // 0) reset barrier counter (workspace is poisoned; must be deterministic)
    init_bar_kernel<<<1, 1, 0, stream>>>(bar);

    // 1) embedding gather
    embed_kernel<<<T1, 256, 0, stream>>>(sentence, embedding, xseq);

    // 2) weight conversion to bf16
    cvt_bf16_kernel<<<1024, 256, 0, stream>>>(fc1_w, w1B, (size_t)D2 * KDIM / 4);
    cvt_bf16_kernel<<<8192, 256, 0, stream>>>(fc2_w, w2B, (size_t)NVOCAB * KDIM / 4);

    // 3) sequential dual-LSTM scan (persistent, grid-synced)
    lstm_scan_kernel<<<SCAN_NBLK, SCAN_TPB, 0, stream>>>(
        xseq, lang_ids, Wih_en, Whh_en, Wih_cn, Whh_cn,
        h_en, h_cn, lstmF, lstmB, bar);

    // 4) fc1 + relu -> bf16 h1
    {
        dim3 grid(D2 / BN, T1 / BM);
        gemm_bf16_kernel<true, true><<<grid, 256, 0, stream>>>(
            lstmB, w1B, fc1_b, nullptr, h1B, T1, D2, KDIM);
    }
    // 5) fc2 -> prediction (f32)
    {
        dim3 grid((NVOCAB + BN - 1) / BN, T1 / BM);
        gemm_bf16_kernel<false, false><<<grid, 256, 0, stream>>>(
            h1B, w2B, fc2_b, pred, nullptr, T1, NVOCAB, KDIM);
    }
    // 6) language head
    lang_head_kernel<<<(T1 * 32 + 255) / 256, 256, 0, stream>>>(lstmF, lang_w, lang_b, lang);
}
